// TransformerBlock_48172353192051
// MI455X (gfx1250) — matrix operand order repaired and checked by reading, not compiled, not run on hardware
//
#include <hip/hip_runtime.h>
#include <hip/hip_bf16.h>
#include <math.h>

typedef __attribute__((ext_vector_type(16))) __bf16 bf16x16;
typedef __attribute__((ext_vector_type(8)))  __bf16 bf16x8;
typedef __attribute__((ext_vector_type(8)))  float  f32x8;
typedef unsigned int uint32;
typedef __attribute__((address_space(3))) char lds_char_t;

static __device__ __forceinline__ bf16x16 cat16(bf16x8 a, bf16x8 b) {
  return __builtin_shufflevector(a, b, 0,1,2,3,4,5,6,7,8,9,10,11,12,13,14,15);
}
static __device__ __forceinline__ f32x8 wmma_bf16(bf16x16 a, bf16x16 b, f32x8 c) {
  // D = A(16x32 bf16) * B(32x16 bf16) + C(16x16 f32)
  return __builtin_amdgcn_wmma_f32_16x16x32_bf16(false, a, false, b, (short)0, c, false, false);
}
// LDS byte offset (relative to workgroup LDS base) of a __shared__ pointer
static __device__ __forceinline__ uint32 lds_addr_of(const void* p) {
  return (uint32)(size_t)(lds_char_t*)p;
}
// gfx1250 async direct memory->LDS copy (16B per lane), tracked by ASYNCcnt
static __device__ __forceinline__ void async_ld_b128(uint32 lds_off, const void* g) {
  asm volatile("global_load_async_to_lds_b128 %0, %1, off"
               :: "v"(lds_off), "v"((unsigned long long)(size_t)g) : "memory");
}
static __device__ __forceinline__ void wait_async() {
  asm volatile("s_wait_asynccnt 0x0" ::: "memory");
}

// ---------------- LayerNorm (fp32 in -> bf16 out), one block per row of 1024 ----
__global__ __launch_bounds__(256) void ln_bf16_kernel(const float* __restrict__ x,
    const float* __restrict__ sc, const float* __restrict__ sh,
    __bf16* __restrict__ out) {
  __shared__ float red[256];
  const int row = blockIdx.x, t = threadIdx.x;
  const float* xr = x + (size_t)row * 1024;
  float v[4]; float s = 0.f;
  #pragma unroll
  for (int i = 0; i < 4; ++i) { v[i] = xr[t + i*256]; s += v[i]; }
  red[t] = s; __syncthreads();
  for (int o = 128; o > 0; o >>= 1) { if (t < o) red[t] += red[t+o]; __syncthreads(); }
  const float mean = red[0] * (1.f/1024.f);
  __syncthreads();
  s = 0.f;
  #pragma unroll
  for (int i = 0; i < 4; ++i) { float d = v[i] - mean; s += d*d; }
  red[t] = s; __syncthreads();
  for (int o = 128; o > 0; o >>= 1) { if (t < o) red[t] += red[t+o]; __syncthreads(); }
  const float rstd = rsqrtf(red[0] * (1.f/1024.f) + 1e-5f);
  __bf16* orow = out + (size_t)row * 1024;
  #pragma unroll
  for (int i = 0; i < 4; ++i) {
    int c = t + i*256;
    orow[c] = (__bf16)(sc[c] * (v[i] - mean) * rstd + sh[c]);
  }
}

// ------------- transposing fp32 -> bf16 weight convert: out[n*K+k] = in[k*N+n] --
__global__ __launch_bounds__(256) void cvt_t_bf16_kernel(const float* __restrict__ w,
    __bf16* __restrict__ o, int K, int nshift) {
  const int i = blockIdx.x * 256 + threadIdx.x;   // over K*N, grid exact
  const int k = i >> nshift, n = i & ((1 << nshift) - 1);
  o[(size_t)n*K + k] = (__bf16)w[i];
}

// ---------------- bf16 WMMA GEMM: C[M,N] = A[M,K] * Bt[N,K]^T (+bias,+gelu,+res)
// 256 threads = 8 waves; block tile 128x128; wave tile 32x64 = 2x4 WMMA tiles.
// Double-buffered LDS tiles filled with global_load_async_to_lds_b128:
//   wait_async -> barrier -> issue async for tile i+1 -> compute tile i.
template<bool BIAS, bool GELU, bool RES, bool OUTBF>
__global__ __launch_bounds__(256) void gemm_bf16_kernel(
    const __bf16* __restrict__ A, const __bf16* __restrict__ Bt,
    const float* __restrict__ bias, const float* __restrict__ res,
    float* __restrict__ outF, __bf16* __restrict__ outB,
    int M, int N, int K) {
  __shared__ __align__(16) __bf16 As[2][128*32];   // [row][k]
  __shared__ __align__(16) __bf16 Bs[2][128*32];   // [n][k]
  const int t = threadIdx.x, lane = t & 31, wid = t >> 5;
  const int ln = lane & 15, hh = lane >> 4;
  const int wm = wid >> 1, wn = wid & 1;
  const int row0 = blockIdx.y * 128, col0 = blockIdx.x * 128;
  const uint32 asb[2] = { lds_addr_of(As[0]), lds_addr_of(As[1]) };
  const uint32 bsb[2] = { lds_addr_of(Bs[0]), lds_addr_of(Bs[1]) };
  auto stage = [&](int buf, int kt) {
    #pragma unroll
    for (int i = 0; i < 2; ++i) {              // 512 x 16B chunks per tile
      int c = t + i*256; int r = c >> 2, k8 = (c & 3) * 8;
      async_ld_b128(asb[buf] + c*16, A  + (size_t)(row0 + r)*K + kt + k8);
      async_ld_b128(bsb[buf] + c*16, Bt + (size_t)(col0 + r)*K + kt + k8);
    }
  };
  f32x8 acc[2][4] = {};
  const int nk = K >> 5;
  stage(0, 0);
  for (int it = 0; it < nk; ++it) {
    const int cur = it & 1;
    wait_async();                               // tile `cur` landed in LDS
    __syncthreads();                            // publish; prev buffer free
    if (it + 1 < nk) stage(cur ^ 1, (it + 1) << 5);   // prefetch under compute
    const __bf16* Ac = As[cur];
    const __bf16* Bc = Bs[cur];
    bf16x16 af[2], bfv[4];
    #pragma unroll
    for (int mi = 0; mi < 2; ++mi) {            // A frag: k = 16h..16h+15 contiguous
      const __bf16* p = &Ac[(wm*32 + mi*16 + ln)*32];
      af[mi] = cat16(*(const bf16x8*)(p + 16*hh), *(const bf16x8*)(p + 16*hh + 8));
    }
    #pragma unroll
    for (int ni = 0; ni < 4; ++ni) {            // B frag: k = 16h..16h+15 contiguous
      const __bf16* p = &Bc[(wn*64 + ni*16 + ln)*32 + 16*hh];
      bfv[ni] = cat16(*(const bf16x8*)p, *(const bf16x8*)(p + 8));
    }
    #pragma unroll
    for (int mi = 0; mi < 2; ++mi)
      #pragma unroll
      for (int ni = 0; ni < 4; ++ni)
        acc[mi][ni] = wmma_bf16(af[mi], bfv[ni], acc[mi][ni]);
  }
  #pragma unroll
  for (int mi = 0; mi < 2; ++mi)
    #pragma unroll
    for (int ni = 0; ni < 4; ++ni)
      #pragma unroll
      for (int g = 0; g < 8; ++g) {
        const int m = row0 + wm*32 + mi*16 + g + 8*hh;   // C layout: M=g+8*half
        const int n = col0 + wn*64 + ni*16 + ln;          // N = lane%16
        float v = acc[mi][ni][g];
        if (BIAS) v += bias[n];
        if (GELU) {
          // 0.5*v*(1+tanh(z)) = v * (1 - 1/(exp(2z)+1)); saturates safely
          float z = 0.7978845608f * (v + 0.044715f * v * v * v);
          float e = __expf(2.f * z);
          v = v * (1.f - 1.f / (e + 1.f));
        }
        if (RES)  v += res[(size_t)m*N + n];
        if (OUTBF) outB[(size_t)m*N + n] = (__bf16)v;
        else       outF[(size_t)m*N + n] = v;
      }
}

// ---------------- causal flash attention, bf16 Q/K/V/ctx --------------------
// grid (qblock=16, head=16, batch=2), 256 threads = 8 waves; wave owns 16 q rows.
__global__ __launch_bounds__(256) void attn_kernel(
    const __bf16* __restrict__ Q, const __bf16* __restrict__ K,
    const __bf16* __restrict__ V, __bf16* __restrict__ ctx) {
  __shared__ __align__(16) __bf16 Ks[128*64];    // [kv][d]
  __shared__ __align__(16) __bf16 Vt[64*128];    // transposed: [d][kv]
  __shared__ __align__(16) __bf16 Ps[8*16*128];  // per-wave P tile [16][128]
  const int t = threadIdx.x, lane = t & 31, wid = t >> 5;
  const int ln = lane & 15, hh = lane >> 4;
  const int qb = blockIdx.x, h = blockIdx.y, b = blockIdx.z;
  const int tok0 = b * 2048, qbase = qb * 128;
  const uint32 ksb = lds_addr_of(Ks);

  bf16x16 qf[2];                                  // Q A-frags, kept in registers
  {
    const __bf16* qp = Q + (size_t)(tok0 + qbase + wid*16 + ln)*1024 + h*64;
    #pragma unroll
    for (int s = 0; s < 2; ++s)
      qf[s] = cat16(*(const bf16x8*)(qp + s*32 + 16*hh),
                    *(const bf16x8*)(qp + s*32 + 16*hh + 8));
  }
  f32x8 O[4] = {};
  float mrow[8], lrow[8];
  #pragma unroll
  for (int g = 0; g < 8; ++g) { mrow[g] = -3.0e38f; lrow[g] = 0.f; }
  __bf16* pw = &Ps[wid * 16 * 128];

  for (int kb = 0; kb <= qb; ++kb) {
    __syncthreads();
    // K block: async direct-to-LDS copy, 1024 x 16B chunks (8 per 64-elem row)
    #pragma unroll
    for (int i = 0; i < 4; ++i) {
      int c = t + i*256; int r = c >> 3, d8 = (c & 7) * 8;
      async_ld_b128(ksb + c*16,
                    K + (size_t)(tok0 + kb*128 + r)*1024 + h*64 + d8);
    }
    // V block transposed: batched vector loads, then scatter stores
    bf16x8 vv[4];
    #pragma unroll
    for (int i = 0; i < 4; ++i) {
      int c = t + i*256; int r = c >> 3, d8 = (c & 7) * 8;
      vv[i] = *(const bf16x8*)(V + (size_t)(tok0 + kb*128 + r)*1024 + h*64 + d8);
    }
    #pragma unroll
    for (int i = 0; i < 4; ++i) {
      int c = t + i*256; int r = c >> 3, d8 = (c & 7) * 8;
      #pragma unroll
      for (int j = 0; j < 8; ++j) Vt[(d8 + j)*128 + r] = vv[i][j];
    }
    wait_async();
    __syncthreads();

    f32x8 S[8] = {};                              // S = Q K^T, 16x128 per wave
    #pragma unroll
    for (int s = 0; s < 2; ++s)
      #pragma unroll
      for (int ni = 0; ni < 8; ++ni) {
        const __bf16* kp = &Ks[(ni*16 + ln)*64 + s*32 + 16*hh];
        S[ni] = wmma_bf16(qf[s],
                          cat16(*(const bf16x8*)kp, *(const bf16x8*)(kp + 8)),
                          S[ni]);
      }
    const bool diag = (kb == qb);
    #pragma unroll
    for (int ni = 0; ni < 8; ++ni)
      #pragma unroll
      for (int g = 0; g < 8; ++g) {
        float v = S[ni][g] * 0.125f;              // 1/sqrt(64)
        if (diag && (ni*16 + ln) > (wid*16 + g + 8*hh)) v = -3.0e38f;
        S[ni][g] = v;
      }
    #pragma unroll
    for (int g = 0; g < 8; ++g) {                 // online softmax per row
      float mx = S[0][g];
      #pragma unroll
      for (int ni = 1; ni < 8; ++ni) mx = fmaxf(mx, S[ni][g]);
      #pragma unroll
      for (int msk = 1; msk < 16; msk <<= 1) mx = fmaxf(mx, __shfl_xor(mx, msk, 32));
      const float mnew = fmaxf(mrow[g], mx);
      const float corr = __expf(mrow[g] - mnew);
      float sum = 0.f;
      #pragma unroll
      for (int ni = 0; ni < 8; ++ni) {
        float p = __expf(S[ni][g] - mnew);
        S[ni][g] = p; sum += p;
      }
      #pragma unroll
      for (int msk = 1; msk < 16; msk <<= 1) sum += __shfl_xor(sum, msk, 32);
      lrow[g] = lrow[g]*corr + sum;
      mrow[g] = mnew;
      #pragma unroll
      for (int ni = 0; ni < 4; ++ni) O[ni][g] *= corr;
    }
    #pragma unroll
    for (int ni = 0; ni < 8; ++ni)                // P: C-layout -> LDS row-major
      #pragma unroll
      for (int g = 0; g < 8; ++g)
        pw[(g + 8*hh)*128 + ni*16 + ln] = (__bf16)S[ni][g];
    asm volatile("s_wait_dscnt 0" ::: "memory");  // same-wave LDS RAW
    #pragma unroll
    for (int s = 0; s < 4; ++s) {                 // O += P * V
      const __bf16* pr = pw + ln*128 + s*32;
      bf16x16 af = cat16(*(const bf16x8*)(pr + 16*hh),
                         *(const bf16x8*)(pr + 16*hh + 8));
      #pragma unroll
      for (int ni = 0; ni < 4; ++ni) {
        const __bf16* vp = &Vt[(ni*16 + ln)*128 + s*32 + 16*hh];
        O[ni] = wmma_bf16(af,
                          cat16(*(const bf16x8*)vp, *(const bf16x8*)(vp + 8)),
                          O[ni]);
      }
    }
  }
  #pragma unroll
  for (int g = 0; g < 8; ++g) {
    const float inv = 1.f / lrow[g];
    const size_t orow = (size_t)(tok0 + qbase + wid*16 + g + 8*hh)*1024 + h*64;
    #pragma unroll
    for (int ni = 0; ni < 4; ++ni)
      ctx[orow + ni*16 + ln] = (__bf16)(O[ni][g] * inv);
  }
}

// ---------------- host orchestration ----------------
extern "C" void kernel_launch(void* const* d_in, const int* in_sizes, int n_in,
                              void* d_out, int out_size, void* d_ws, size_t ws_size,
                              hipStream_t stream) {
  const float* x    = (const float*)d_in[0];
  const float* Wq   = (const float*)d_in[1];
  const float* Wk   = (const float*)d_in[2];
  const float* Wv   = (const float*)d_in[3];
  const float* Wo   = (const float*)d_in[4];
  const float* bo   = (const float*)d_in[5];
  const float* W1   = (const float*)d_in[6];
  const float* b1   = (const float*)d_in[7];
  const float* W2   = (const float*)d_in[8];
  const float* b2   = (const float*)d_in[9];
  const float* ln1s = (const float*)d_in[10];
  const float* ln1b = (const float*)d_in[11];
  const float* ln2s = (const float*)d_in[12];
  const float* ln2b = (const float*)d_in[13];
  float* out = (float*)d_out;                     // doubles as x1 residual buffer

  char* ws = (char*)d_ws;
  __bf16* hbf = (__bf16*)(ws);                    //  8 MB  LN output (bf16)
  __bf16* wbf = (__bf16*)(ws + ( 8u<<20));        //  8 MB  transposed bf16 weights
  __bf16* qbf = (__bf16*)(ws + (16u<<20));        //  8 MB
  __bf16* kbf = (__bf16*)(ws + (24u<<20));        //  8 MB
  __bf16* vbf = (__bf16*)(ws + (32u<<20));        //  8 MB
  __bf16* cbf = (__bf16*)(ws + (40u<<20));        //  8 MB  attention context
  __bf16* fbf = (__bf16*)(ws + (48u<<20));        // 32 MB  FFN hidden

  const int TOK = 4096, D = 1024, F = 4096;
  const dim3 blk(256);
  const dim3 gD(D/128, TOK/128);                  // N=1024 GEMMs
  const dim3 gF(F/128, TOK/128);                  // N=4096 GEMM

  // LN1
  ln_bf16_kernel<<<TOK, blk, 0, stream>>>(x, ln1s, ln1b, hbf);
  // Q,K,V projections (weights converted+transposed to [N][K] bf16)
  cvt_t_bf16_kernel<<<(D*D)/256, blk, 0, stream>>>(Wq, wbf, D, 10);
  gemm_bf16_kernel<false,false,false,true><<<gD, blk, 0, stream>>>(
      hbf, wbf, nullptr, nullptr, nullptr, qbf, TOK, D, D);
  cvt_t_bf16_kernel<<<(D*D)/256, blk, 0, stream>>>(Wk, wbf, D, 10);
  gemm_bf16_kernel<false,false,false,true><<<gD, blk, 0, stream>>>(
      hbf, wbf, nullptr, nullptr, nullptr, kbf, TOK, D, D);
  cvt_t_bf16_kernel<<<(D*D)/256, blk, 0, stream>>>(Wv, wbf, D, 10);
  gemm_bf16_kernel<false,false,false,true><<<gD, blk, 0, stream>>>(
      hbf, wbf, nullptr, nullptr, nullptr, vbf, TOK, D, D);
  // causal flash attention
  attn_kernel<<<dim3(16,16,2), blk, 0, stream>>>(qbf, kbf, vbf, cbf);
  // output projection + bias + residual(x) -> x1 in d_out (fp32)
  cvt_t_bf16_kernel<<<(D*D)/256, blk, 0, stream>>>(Wo, wbf, D, 10);
  gemm_bf16_kernel<true,false,true,false><<<gD, blk, 0, stream>>>(
      cbf, wbf, bo, x, out, nullptr, TOK, D, D);
  // LN2
  ln_bf16_kernel<<<TOK, blk, 0, stream>>>(out, ln2s, ln2b, hbf);
  // FFN1: gelu(h @ W1 + b1) -> bf16   (W1 [1024,4096] -> Wt [4096][1024])
  cvt_t_bf16_kernel<<<(D*F)/256, blk, 0, stream>>>(W1, wbf, D, 12);
  gemm_bf16_kernel<true,true,false,true><<<gF, blk, 0, stream>>>(
      hbf, wbf, b1, nullptr, nullptr, fbf, TOK, F, D);
  // FFN2: (+ b2 + x1) -> d_out fp32   (W2 [4096,1024] -> Wt [1024][4096])
  cvt_t_bf16_kernel<<<(F*D)/256, blk, 0, stream>>>(W2, wbf, F, 10);
  gemm_bf16_kernel<true,false,true,false><<<gD, blk, 0, stream>>>(
      fbf, wbf, b2, out, out, nullptr, TOK, D, F);
}
